// DistSelfAttention_14276471292156
// MI455X (gfx1250) — matrix operand order repaired and checked by reading, not compiled, not run on hardware
//
#include <hip/hip_runtime.h>

// ---------------------------------------------------------------------------
// Distributional (Wasserstein) self-attention for MI455X (gfx1250, wave32).
// bf16 WMMA (f32 accumulate); async-to-LDS weight staging; fused epilogues.
// ---------------------------------------------------------------------------

constexpr int B   = 64;
constexpr int S   = 512;
constexpr int HID = 256;
constexpr int NH  = 4;
constexpr int DH  = 64;
constexpr int M   = B * S;          // 32768 rows
constexpr int BH  = B * NH;         // 256 batched heads

typedef __attribute__((ext_vector_type(16))) __bf16 v16bf;
typedef __attribute__((ext_vector_type(2)))  __bf16 v2bf;
typedef __attribute__((ext_vector_type(8)))  float  v8f;

#define DEVINL __device__ __forceinline__

#ifndef __has_builtin
#define __has_builtin(x) 0
#endif
#if __has_builtin(__builtin_amdgcn_global_load_async_to_lds_b128)
#define HAS_ASYNC_LDS 1
#else
#define HAS_ASYNC_LDS 0
#endif
#if __has_builtin(__builtin_amdgcn_cvt_pk_bf16_f32)
#define HAS_CVT_PK 1
#else
#define HAS_CVT_PK 0
#endif
#if __has_builtin(__builtin_amdgcn_s_wait_asynccnt)
#define HAS_WAIT_ASYNC 1
#else
#define HAS_WAIT_ASYNC 0
#endif

union FragU { v16bf v; uint4 q[2]; unsigned u[8]; };

DEVINL unsigned short bf16bits(float f) {
  unsigned u = __builtin_bit_cast(unsigned, f);
  return (unsigned short)((u + 0x7FFFu + ((u >> 16) & 1u)) >> 16);  // RNE
}
DEVINL unsigned pk(float a, float b) {
#if HAS_CVT_PK
  v2bf r = __builtin_amdgcn_cvt_pk_bf16_f32(a, b);
  return __builtin_bit_cast(unsigned, r);
#else
  return (unsigned)bf16bits(a) | ((unsigned)bf16bits(b) << 16);
#endif
}
DEVINL __bf16 f2bf(float f) { return __builtin_bit_cast(__bf16, bf16bits(f)); }

// acc += lo(u)^2 + hi(u)^2 where u packs two bf16
DEVINL float acc2(unsigned u, float acc) {
  float lo = __builtin_bit_cast(float, u << 16);
  float hi = __builtin_bit_cast(float, u & 0xFFFF0000u);
  return acc + lo * lo + hi * hi;
}

DEVINL v8f wmma_bf16(v16bf a, v16bf b, v8f c) {
  return __builtin_amdgcn_wmma_f32_16x16x32_bf16(false, a, false, b, (short)0, c,
                                                 false, false);
}

// ------------------------- async LDS staging helpers -----------------------
// Builtin signature (from hipcc diagnostic): pointee must be the GCC-style
// 'int __vector(4)' (vector_size(16)), global (AS1) src + LDS (AS3) dst.
typedef int gv4i __attribute__((vector_size(16)));
typedef __attribute__((address_space(1))) gv4i* as1_v4p;
typedef __attribute__((address_space(3))) gv4i* as3_v4p;

DEVINL void async_copy16(const void* g, void* l) {
#if HAS_ASYNC_LDS
  __builtin_amdgcn_global_load_async_to_lds_b128((as1_v4p)g, (as3_v4p)l, 0, 0);
#else
  // inline-asm path per CDNA5 ISA (VGLOBAL encoding):
  // VDST = 32-bit LDS byte address, VADDR = 64-bit global address.
  unsigned lds_off = (unsigned)(__UINTPTR_TYPE__)(as3_v4p)l;
  unsigned long long ga = (unsigned long long)(__UINTPTR_TYPE__)g;
  asm volatile("global_load_async_to_lds_b128 %0, %1, off"
               :
               : "v"(lds_off), "v"(ga)
               : "memory");
#endif
}

DEVINL void wait_async0() {
#if HAS_WAIT_ASYNC
  __builtin_amdgcn_s_wait_asynccnt(0);
#else
  asm volatile("s_wait_asynccnt 0x0" ::: "memory");
#endif
}

// ------------------------------ fragment loads -----------------------------
// A-frag (16x32, M x K), 16-bit layout: lane<16 -> K {0..7,16..23}; lane>=16 -> +8.
DEVINL v16bf load_a_f32(const float* __restrict__ src, int ld, int row0, int kc,
                        int lane) {
  const float* p = src + (long)(row0 + (lane & 15)) * ld + kc + ((lane & 16) ? 8 : 0);
  float4 c0 = *(const float4*)(p);
  float4 c1 = *(const float4*)(p + 4);
  float4 c2 = *(const float4*)(p + 16);
  float4 c3 = *(const float4*)(p + 20);
  FragU f;
  f.u[0] = pk(c0.x, c0.y); f.u[1] = pk(c0.z, c0.w);
  f.u[2] = pk(c1.x, c1.y); f.u[3] = pk(c1.z, c1.w);
  f.u[4] = pk(c2.x, c2.y); f.u[5] = pk(c2.z, c2.w);
  f.u[6] = pk(c3.x, c3.y); f.u[7] = pk(c3.z, c3.w);
  return f.v;
}

// Same, but also returns element-wise squared frag (for probs^2 @ cv).
DEVINL void load_a_f32_sq(const float* __restrict__ src, int ld, int row0, int kc,
                          int lane, v16bf& a, v16bf& a2) {
  const float* p = src + (long)(row0 + (lane & 15)) * ld + kc + ((lane & 16) ? 8 : 0);
  float4 c0 = *(const float4*)(p);
  float4 c1 = *(const float4*)(p + 4);
  float4 c2 = *(const float4*)(p + 16);
  float4 c3 = *(const float4*)(p + 20);
  FragU f, g;
  f.u[0] = pk(c0.x, c0.y); f.u[1] = pk(c0.z, c0.w);
  f.u[2] = pk(c1.x, c1.y); f.u[3] = pk(c1.z, c1.w);
  f.u[4] = pk(c2.x, c2.y); f.u[5] = pk(c2.z, c2.w);
  f.u[6] = pk(c3.x, c3.y); f.u[7] = pk(c3.z, c3.w);
  g.u[0] = pk(c0.x*c0.x, c0.y*c0.y); g.u[1] = pk(c0.z*c0.z, c0.w*c0.w);
  g.u[2] = pk(c1.x*c1.x, c1.y*c1.y); g.u[3] = pk(c1.z*c1.z, c1.w*c1.w);
  g.u[4] = pk(c2.x*c2.x, c2.y*c2.y); g.u[5] = pk(c2.z*c2.z, c2.w*c2.w);
  g.u[6] = pk(c3.x*c3.x, c3.y*c3.y); g.u[7] = pk(c3.z*c3.z, c3.w*c3.w);
  a = f.v; a2 = g.v;
}

DEVINL v16bf load_a_bf16(const __bf16* __restrict__ src, int ld, int row0, int kc,
                         int lane) {
  const __bf16* p = src + (long)(row0 + (lane & 15)) * ld + kc + ((lane & 16) ? 8 : 0);
  FragU f;
  f.q[0] = *(const uint4*)(p);
  f.q[1] = *(const uint4*)(p + 16);
  return f.v;
}

// B-frag (32x16, K x N): lane n holds column n; lane<16 -> K {0..7,16..23}; lane>=16 -> +8.
// src is column-major-B = [n][k] row-major, k contiguous.
DEVINL v16bf load_b_bf16(const __bf16* __restrict__ src, int ld, int n0, int kc,
                         int lane) {
  const __bf16* p = src + (long)(n0 + (lane & 15)) * ld + kc + ((lane & 16) ? 8 : 0);
  FragU f;
  f.q[0] = *(const uint4*)(p);
  f.q[1] = *(const uint4*)(p + 16);
  return f.v;
}

// ---------------------------------------------------------------------------
// Kernel 0: one-time weight conversion f32 -> bf16 (keeps hot loops cvt-free;
// 1 MB of bf16 weights then lives in the 192 MB L2 for the whole pipeline).
// ---------------------------------------------------------------------------
__global__ __launch_bounds__(256) void w_to_bf16(const float* __restrict__ W,
                                                 __bf16* __restrict__ Wb) {
  int idx = (blockIdx.x * 256 + threadIdx.x) * 4;
  float4 v = *(const float4*)(W + idx);
  uint2 o;
  o.x = pk(v.x, v.y);
  o.y = pk(v.z, v.w);
  *(uint2*)(Wb + idx) = o;
}

// ---------------------------------------------------------------------------
// Kernel 1: projection GEMM  y[m,o] = X[m,:] . W[o,:] + b[o], fused epilogue.
//   ACT: 0=none, 1=elu+1, 2=elu+1 then sqrt.  TR: store [b,h,d,s] else [b,h,s,d].
// WG = 8 waves covering 64M x 256N; W tiles (bf16) staged into LDS with
// double-buffered GLOBAL_LOAD_ASYNC_TO_LDS_B128 (ASYNCcnt-tracked).
// ---------------------------------------------------------------------------
constexpr int WPAD = 40;  // LDS row stride (bf16) for W tiles

template <int ACT, bool TR>
__global__ __launch_bounds__(256) void proj_gemm(const float* __restrict__ X,
                                                 const __bf16* __restrict__ Wb,
                                                 const float* __restrict__ bias,
                                                 __bf16* __restrict__ out) {
  __shared__ __bf16 ldsW[2][256 * WPAD];
  const int wave = threadIdx.x >> 5, lane = threadIdx.x & 31;
  const int m0 = blockIdx.x * 64 + (wave >> 1) * 16;
  const int n0w = (wave & 1) * 128;

  auto stage = [&](int kc, int buf) {
    int o = threadIdx.x;
    const __bf16* src = Wb + o * HID + kc;
    __bf16* dst = &ldsW[buf][o * WPAD];
#pragma unroll
    for (int j = 0; j < 4; j++) async_copy16(src + j * 8, dst + j * 8);
  };

  v8f acc[8] = {};
  stage(0, 0);
  int buf = 0;
  for (int kc = 0; kc < HID; kc += 32) {
    wait_async0();
    __syncthreads();
    if (kc + 32 < HID) stage(kc + 32, buf ^ 1);
    v16bf a = load_a_f32(X, HID, m0, kc, lane);
#pragma unroll
    for (int t = 0; t < 8; t++) {
      int n = n0w + t * 16 + (lane & 15);
      const __bf16* p = &ldsW[buf][n * WPAD + ((lane & 16) ? 8 : 0)];
      FragU bf;
      bf.q[0] = *(const uint4*)(p);
      bf.q[1] = *(const uint4*)(p + 16);
      acc[t] = wmma_bf16(a, bf.v, acc[t]);
    }
    buf ^= 1;
  }
  const int rofs = (lane & 16) ? 8 : 0;
#pragma unroll
  for (int t = 0; t < 8; t++) {
    int o = n0w + t * 16 + (lane & 15);
    float bv = bias[o];
    int h = o >> 6, d = o & 63;
#pragma unroll
    for (int i = 0; i < 8; i++) {
      int m = m0 + i + rofs;
      float x = acc[t][i] + bv;
      if (ACT >= 1) x = (x > 0.f) ? x + 1.f : __expf(x);  // elu(x)+1
      if (ACT == 2) x = sqrtf(x);
      int bi = m >> 9, s = m & (S - 1);
      long off = TR ? ((long)((bi * NH + h) * DH + d) * S + s)
                    : ((long)((bi * NH + h) * S + s) * DH + d);
      out[off] = f2bf(x);
    }
  }
}

// ---------------------------------------------------------------------------
// Kernel 2: per-row norms  n[bh,s] = ||m||^2 + sum(c) = sum(m^2) + sum(sq^2)
// ---------------------------------------------------------------------------
__global__ __launch_bounds__(256) void row_norms(const __bf16* __restrict__ mm,
                                                 const __bf16* __restrict__ ss,
                                                 float* __restrict__ nn) {
  long r = (long)blockIdx.x * 256 + threadIdx.x;  // r < BH*S
  const uint4* a = (const uint4*)(mm + r * DH);
  const uint4* b = (const uint4*)(ss + r * DH);
  float acc = 0.f;
#pragma unroll
  for (int j = 0; j < 8; j++) {
    uint4 va = a[j], vb = b[j];
    acc = acc2(va.x, acc); acc = acc2(va.y, acc);
    acc = acc2(va.z, acc); acc = acc2(va.w, acc);
    acc = acc2(vb.x, acc); acc = acc2(vb.y, acc);
    acc = acc2(vb.z, acc); acc = acc2(vb.w, acc);
  }
  nn[r] = acc;
}

// ---------------------------------------------------------------------------
// Kernel 3: fused scores + softmax.
// WG = one (b,h), one 16-row q tile; wave w owns k in [64w, 64w+64).
// G = mq.mk + sq.sk (WMMA, K=64);  dist = nq+nk-2G;
// w = exp(-dist)/8 + mask;  probs = softmax_k(w)  (double exp, per reference).
// ---------------------------------------------------------------------------
__global__ __launch_bounds__(256) void attn_scores(
    const __bf16* __restrict__ mq, const __bf16* __restrict__ mk,
    const __bf16* __restrict__ sq, const __bf16* __restrict__ sk,
    const float* __restrict__ nq, const float* __restrict__ nk,
    const float* __restrict__ mask, float* __restrict__ probs) {
  __shared__ float red[8][16];
  const int wave = threadIdx.x >> 5, lane = threadIdx.x & 31;
  const int bh = blockIdx.y;
  const int q0 = blockIdx.x * 16;
  const int k0 = wave * 64;
  const long hb = (long)bh * S * DH;

  v16bf amq[2], asq[2];
  amq[0] = load_a_bf16(mq + hb, DH, q0, 0, lane);
  amq[1] = load_a_bf16(mq + hb, DH, q0, 32, lane);
  asq[0] = load_a_bf16(sq + hb, DH, q0, 0, lane);
  asq[1] = load_a_bf16(sq + hb, DH, q0, 32, lane);

  v8f acc[4] = {};
#pragma unroll
  for (int t = 0; t < 4; t++) {
#pragma unroll
    for (int dc = 0; dc < 2; dc++) {
      v16bf bm = load_b_bf16(mk + hb, DH, k0 + t * 16, dc * 32, lane);
      acc[t] = wmma_bf16(amq[dc], bm, acc[t]);
      v16bf bs = load_b_bf16(sk + hb, DH, k0 + t * 16, dc * 32, lane);
      acc[t] = wmma_bf16(asq[dc], bs, acc[t]);
    }
  }

  const int b = bh >> 2;  // NH = 4
  const float* nqp = nq + (long)bh * S;
  const float* nkp = nk + (long)bh * S;
  const float* mkp = mask + (long)b * S * S;
  const int rofs = (lane & 16) ? 8 : 0;

  float w[4][8];
#pragma unroll
  for (int t = 0; t < 4; t++) {
    int k = k0 + t * 16 + (lane & 15);
    float nkv = nkp[k];
#pragma unroll
    for (int i = 0; i < 8; i++) {
      int q = q0 + i + rofs;
      float dist = nqp[q] + nkv - 2.f * acc[t][i];
      w[t][i] = __expf(-dist) * 0.125f + mkp[(long)q * S + k];
    }
  }

  // --- row max over all 512 columns ---
  float pm[8];
#pragma unroll
  for (int i = 0; i < 8; i++) {
    float v = fmaxf(fmaxf(w[0][i], w[1][i]), fmaxf(w[2][i], w[3][i]));
    v = fmaxf(v, __shfl_xor(v, 1, 16));
    v = fmaxf(v, __shfl_xor(v, 2, 16));
    v = fmaxf(v, __shfl_xor(v, 4, 16));
    v = fmaxf(v, __shfl_xor(v, 8, 16));
    pm[i] = v;
  }
  if ((lane & 15) == 0)
    for (int i = 0; i < 8; i++) red[wave][i + rofs] = pm[i];
  __syncthreads();
  float rm[8];
#pragma unroll
  for (int i = 0; i < 8; i++) {
    float v = red[0][i + rofs];
    for (int wv = 1; wv < 8; wv++) v = fmaxf(v, red[wv][i + rofs]);
    rm[i] = v;
  }
  __syncthreads();

  // --- exp + row sum ---
  float ps[8];
#pragma unroll
  for (int i = 0; i < 8; i++) {
    float s = 0.f;
#pragma unroll
    for (int t = 0; t < 4; t++) {
      float e = __expf(w[t][i] - rm[i]);
      w[t][i] = e;
      s += e;
    }
    s += __shfl_xor(s, 1, 16);
    s += __shfl_xor(s, 2, 16);
    s += __shfl_xor(s, 4, 16);
    s += __shfl_xor(s, 8, 16);
    ps[i] = s;
  }
  if ((lane & 15) == 0)
    for (int i = 0; i < 8; i++) red[wave][i + rofs] = ps[i];
  __syncthreads();

  float* pout = probs + (long)bh * S * S;
#pragma unroll
  for (int i = 0; i < 8; i++) {
    float s = 0.f;
    for (int wv = 0; wv < 8; wv++) s += red[wv][i + rofs];
    float inv = 1.f / s;
    int q = q0 + i + rofs;
#pragma unroll
    for (int t = 0; t < 4; t++) {
      int k = k0 + t * 16 + (lane & 15);
      pout[(long)q * S + k] = w[t][i] * inv;
    }
  }
}

// ---------------------------------------------------------------------------
// Kernel 4: context GEMMs  mean_ctx = probs @ mv,  cov_ctx = probs^2 @ cv.
// Wave = 16q x 64d for BOTH outputs; WG = 128q. mv/cv stored [b,h,d,s].
// ---------------------------------------------------------------------------
__global__ __launch_bounds__(256) void ctx_gemm(const float* __restrict__ probs,
                                                const __bf16* __restrict__ mvT,
                                                const __bf16* __restrict__ cvT,
                                                __bf16* __restrict__ mctx,
                                                __bf16* __restrict__ cctx) {
  const int wave = threadIdx.x >> 5, lane = threadIdx.x & 31;
  const int bh = blockIdx.y, b = bh >> 2, h = bh & 3;
  const int q0 = blockIdx.x * 128 + wave * 16;
  const float* P = probs + (long)bh * S * S;
  const __bf16* MV = mvT + (long)bh * DH * S;
  const __bf16* CV = cvT + (long)bh * DH * S;

  v8f am[4] = {}, ac[4] = {};
  for (int kc = 0; kc < S; kc += 32) {
    if (kc + 32 < S)  // prefetch next probs chunk (-> global_prefetch_b8)
      __builtin_prefetch(P + (long)(q0 + (lane & 15)) * S + kc + 32, 0, 1);
    v16bf a1, a2;
    load_a_f32_sq(P, S, q0, kc, lane, a1, a2);
#pragma unroll
    for (int t = 0; t < 4; t++) {
      v16bf bm = load_b_bf16(MV, S, t * 16, kc, lane);
      am[t] = wmma_bf16(a1, bm, am[t]);
      v16bf bc = load_b_bf16(CV, S, t * 16, kc, lane);
      ac[t] = wmma_bf16(a2, bc, ac[t]);
    }
  }
  const int rofs = (lane & 16) ? 8 : 0;
#pragma unroll
  for (int t = 0; t < 4; t++) {
    int o = h * DH + t * 16 + (lane & 15);
#pragma unroll
    for (int i = 0; i < 8; i++) {
      int s = q0 + i + rofs;
      long off = (long)(b * S + s) * HID + o;
      mctx[off] = f2bf(am[t][i]);
      cctx[off] = f2bf(ac[t][i]);
    }
  }
}

// ---------------------------------------------------------------------------
// Kernel 5: output dense + residual + LayerNorm (fused).
// Same tiling as proj_gemm; row of 256 spans 2 waves -> LDS reduction for LN.
// ---------------------------------------------------------------------------
__global__ __launch_bounds__(256) void dense_ln(const __bf16* __restrict__ Xctx,
                                                const __bf16* __restrict__ Wb,
                                                const float* __restrict__ bias,
                                                const float* __restrict__ resid,
                                                const float* __restrict__ lnw,
                                                const float* __restrict__ lnb,
                                                float* __restrict__ out) {
  __shared__ __bf16 ldsW[2][256 * WPAD];
  __shared__ float redS[4][16][2][2];  // [mblk][row][nhalf][{sum,sumsq}]
  const int wave = threadIdx.x >> 5, lane = threadIdx.x & 31;
  const int mblk = wave >> 1, nhalf = wave & 1;
  const int m0 = blockIdx.x * 64 + mblk * 16;
  const int n0w = nhalf * 128;

  auto stage = [&](int kc, int buf) {
    int o = threadIdx.x;
    const __bf16* src = Wb + o * HID + kc;
    __bf16* dst = &ldsW[buf][o * WPAD];
#pragma unroll
    for (int j = 0; j < 4; j++) async_copy16(src + j * 8, dst + j * 8);
  };

  v8f acc[8] = {};
  stage(0, 0);
  int buf = 0;
  for (int kc = 0; kc < HID; kc += 32) {
    wait_async0();
    __syncthreads();
    if (kc + 32 < HID) stage(kc + 32, buf ^ 1);
    v16bf a = load_a_bf16(Xctx, HID, m0, kc, lane);
#pragma unroll
    for (int t = 0; t < 8; t++) {
      int n = n0w + t * 16 + (lane & 15);
      const __bf16* p = &ldsW[buf][n * WPAD + ((lane & 16) ? 8 : 0)];
      FragU bf;
      bf.q[0] = *(const uint4*)(p);
      bf.q[1] = *(const uint4*)(p + 16);
      acc[t] = wmma_bf16(a, bf.v, acc[t]);
    }
    buf ^= 1;
  }
  const int rofs = (lane & 16) ? 8 : 0;
  // x = gemm + bias + residual
#pragma unroll
  for (int t = 0; t < 8; t++) {
    int o = n0w + t * 16 + (lane & 15);
    float bv = bias[o];
#pragma unroll
    for (int i = 0; i < 8; i++) {
      long m = m0 + i + rofs;
      acc[t][i] = acc[t][i] + bv + resid[m * HID + o];
    }
  }
  // LN row statistics
  float ps[8], pq[8];
#pragma unroll
  for (int i = 0; i < 8; i++) {
    float s = 0.f, q = 0.f;
#pragma unroll
    for (int t = 0; t < 8; t++) {
      float x = acc[t][i];
      s += x;
      q += x * x;
    }
    s += __shfl_xor(s, 1, 16); q += __shfl_xor(q, 1, 16);
    s += __shfl_xor(s, 2, 16); q += __shfl_xor(q, 2, 16);
    s += __shfl_xor(s, 4, 16); q += __shfl_xor(q, 4, 16);
    s += __shfl_xor(s, 8, 16); q += __shfl_xor(q, 8, 16);
    ps[i] = s; pq[i] = q;
  }
  if ((lane & 15) == 0) {
#pragma unroll
    for (int i = 0; i < 8; i++) {
      redS[mblk][i + rofs][nhalf][0] = ps[i];
      redS[mblk][i + rofs][nhalf][1] = pq[i];
    }
  }
  __syncthreads();
  float um[8], inv[8];
#pragma unroll
  for (int i = 0; i < 8; i++) {
    int r = i + rofs;
    float s = redS[mblk][r][0][0] + redS[mblk][r][1][0];
    float q = redS[mblk][r][0][1] + redS[mblk][r][1][1];
    float u = s * (1.f / HID);
    float var = q * (1.f / HID) - u * u;
    um[i] = u;
    inv[i] = rsqrtf(var + 1e-12f);
  }
#pragma unroll
  for (int t = 0; t < 8; t++) {
    int o = n0w + t * 16 + (lane & 15);
    float gw = lnw[o], gb = lnb[o];
#pragma unroll
    for (int i = 0; i < 8; i++) {
      long m = m0 + i + rofs;
      out[m * HID + o] = gw * (acc[t][i] - um[i]) * inv[i] + gb;
    }
  }
}

// ---------------------------------------------------------------------------
// Host launcher
// ---------------------------------------------------------------------------
extern "C" void kernel_launch(void* const* d_in, const int* in_sizes, int n_in,
                              void* d_out, int out_size, void* d_ws, size_t ws_size,
                              hipStream_t stream) {
  (void)in_sizes; (void)n_in; (void)out_size; (void)ws_size;
  const float* in_mean = (const float*)d_in[0];
  const float* in_cov  = (const float*)d_in[1];
  const float* mask    = (const float*)d_in[2];
  const float* Wf[8]   = {(const float*)d_in[3],  (const float*)d_in[5],
                          (const float*)d_in[7],  (const float*)d_in[9],
                          (const float*)d_in[11], (const float*)d_in[13],
                          (const float*)d_in[15], (const float*)d_in[17]};
  const float* bmq = (const float*)d_in[4],  *bmk = (const float*)d_in[6];
  const float* bmv = (const float*)d_in[8],  *bcq = (const float*)d_in[10];
  const float* bck = (const float*)d_in[12], *bcv = (const float*)d_in[14];
  const float* bmd = (const float*)d_in[16], *bcd = (const float*)d_in[18];
  const float* lnw = (const float*)d_in[19], *lnb = (const float*)d_in[20];

  float* out_mean = (float*)d_out;
  float* out_cov  = out_mean + (size_t)M * HID;
  float* probs    = out_cov + (size_t)M * HID;

  const size_t TEN = (size_t)BH * S * DH;  // elems per head tensor
  char* ws = (char*)d_ws;
  __bf16* mq  = (__bf16*)ws; ws += TEN * 2;
  __bf16* mk  = (__bf16*)ws; ws += TEN * 2;
  __bf16* sq  = (__bf16*)ws; ws += TEN * 2;
  __bf16* sk  = (__bf16*)ws; ws += TEN * 2;
  __bf16* mvT = (__bf16*)ws; ws += TEN * 2;
  __bf16* cvT = (__bf16*)ws; ws += TEN * 2;
  __bf16* mctx = (__bf16*)ws; ws += TEN * 2;
  __bf16* cctx = (__bf16*)ws; ws += TEN * 2;
  float* nq = (float*)ws; ws += (size_t)BH * S * 4;
  float* nk = (float*)ws; ws += (size_t)BH * S * 4;
  __bf16* Wb[8];
  for (int i = 0; i < 8; i++) { Wb[i] = (__bf16*)ws; ws += (size_t)HID * HID * 2; }

  dim3 blk(256);
  // 0) weight conversion (once; bf16 weights stay L2-resident)
  for (int i = 0; i < 8; i++)
    w_to_bf16<<<(HID * HID) / 1024, blk, 0, stream>>>(Wf[i], Wb[i]);
  // 1) projections (fused bias / elu+1 / sqrt / head-layout)
  proj_gemm<0, false><<<M / 64, blk, 0, stream>>>(in_mean, Wb[0], bmq, mq);
  proj_gemm<0, false><<<M / 64, blk, 0, stream>>>(in_mean, Wb[1], bmk, mk);
  proj_gemm<0, true ><<<M / 64, blk, 0, stream>>>(in_mean, Wb[2], bmv, mvT);
  proj_gemm<2, false><<<M / 64, blk, 0, stream>>>(in_cov,  Wb[3], bcq, sq);
  proj_gemm<2, false><<<M / 64, blk, 0, stream>>>(in_cov,  Wb[4], bck, sk);
  proj_gemm<1, true ><<<M / 64, blk, 0, stream>>>(in_cov,  Wb[5], bcv, cvT);
  // 2) norm terms
  row_norms<<<(BH * S) / 256, blk, 0, stream>>>(mq, sq, nq);
  row_norms<<<(BH * S) / 256, blk, 0, stream>>>(mk, sk, nk);
  // 3) fused scores + softmax -> probs (output, reused as ctx input)
  attn_scores<<<dim3(S / 16, BH), blk, 0, stream>>>(mq, mk, sq, sk, nq, nk, mask,
                                                    probs);
  // 4) context GEMMs (probs and probs^2)
  ctx_gemm<<<dim3(S / 128, BH), blk, 0, stream>>>(probs, mvT, cvT, mctx, cctx);
  // 5) output dense + residual + LayerNorm
  dense_ln<<<M / 64, blk, 0, stream>>>(mctx, Wb[6], bmd, in_mean, lnw, lnb, out_mean);
  dense_ln<<<M / 64, blk, 0, stream>>>(cctx, Wb[7], bcd, in_cov,  lnw, lnb, out_cov);
}
